// GatedMultiheadAttention_28733331210508
// MI455X (gfx1250) — matrix operand order repaired and checked by reading, not compiled, not run on hardware
//
#include <hip/hip_runtime.h>
#include <math.h>

// ---------------------------------------------------------------------------
// Gated multi-head attention for MI455X (gfx1250), bf16 WMMA path.
//   B=2, L=2048, C=1024, H=16, D=64
// Roofline: ~77 GFLOP, dominant HBM traffic = 536MB bias (streamed NT,
// software-pipelined by one kv tile). All GEMM operands stay L2-resident
// (<=16MB each vs 192MB L2); weights pre-packed into exact v_wmma B-fragment
// layout so every operand load is one contiguous 32B (or 2x16B) per lane.
// Softmax cross-lane reductions use v_permlane16_b32 (VALU) instead of
// ds_bpermute to keep the LDS unit free and avoid dscnt stalls.
// ---------------------------------------------------------------------------

typedef __bf16 bf16;
typedef __attribute__((ext_vector_type(16))) __bf16 v16bf;
typedef __attribute__((ext_vector_type(8)))  __bf16 v8bf;
typedef __attribute__((ext_vector_type(8)))  float  v8f;

#define B_  2
#define L_  2048
#define C_  1024
#define H_  16
#define D_  64
#define M_  (B_ * L_)          // 4096 rows
#define LOG2E 1.44269504088896340736f

union ABf { v16bf v; v8bf h[2]; };

static __device__ __forceinline__ v8f wmma_bf16(v16bf a, v16bf b, v8f c) {
  // 8-arg form: (neg_a, A, neg_b, B, c_mod, C, reuse_a, reuse_b)
  return __builtin_amdgcn_wmma_f32_16x16x32_bf16(false, a, false, b,
                                                 (short)0, c, false, false);
}

// xor-butterfly across 16-lane rows via v_permlane16_b32 (no LDS traffic).
static __device__ __forceinline__ float pl16(float x, unsigned s1, unsigned s2) {
  int b = __float_as_int(x);
  return __int_as_float(
      __builtin_amdgcn_permlane16(b, b, (int)s1, (int)s2, false, false));
}
static __device__ __forceinline__ float xmax16(float v) {
  v = fmaxf(v, pl16(v, 0x67452301u, 0xEFCDAB89u));  // xor 1
  v = fmaxf(v, pl16(v, 0x54761032u, 0xDCFE98BAu));  // xor 2
  v = fmaxf(v, pl16(v, 0x32107654u, 0xBA98FEDCu));  // xor 4
  v = fmaxf(v, pl16(v, 0xFEDCBA98u, 0x76543210u));  // xor 8
  return v;
}
static __device__ __forceinline__ float xsum16(float v) {
  v += pl16(v, 0x67452301u, 0xEFCDAB89u);
  v += pl16(v, 0x54761032u, 0xDCFE98BAu);
  v += pl16(v, 0x32107654u, 0xBA98FEDCu);
  v += pl16(v, 0xFEDCBA98u, 0x76543210u);
  return v;
}

// --------------------------- f32 -> bf16 convert ---------------------------
__global__ void k_cvt(const float* __restrict__ x, bf16* __restrict__ y, int n) {
  int i = blockIdx.x * 256 + threadIdx.x;
  if (i < n) y[i] = (bf16)x[i];
}

// ------------------- pack weight [K=1024][N=1024] into ---------------------
// B-fragment layout Wp[nt=64][kt=32][lane=32][j=16]:
//   lane: col = lane&15, k-chunk = (lane>>4)*16 ; element j <-> k = chunk+j
__global__ void k_packw(const float* __restrict__ w, bf16* __restrict__ wp) {
  int i = blockIdx.x * 256 + threadIdx.x;       // 1M elements
  int j    = i & 15;
  int lane = (i >> 4) & 31;
  int kt   = (i >> 9) & 31;
  int nt   = i >> 14;
  int k = kt * 32 + ((lane >> 4) << 4) + j;
  int n = nt * 16 + (lane & 15);
  wp[i] = (bf16)w[(size_t)k * C_ + n];
}

// ------------------------------ WMMA GEMM ----------------------------------
// Out[m,n] = A[m,:] @ W[:,n] ; block = 8 waves, wave tile 32x64 (2x4 tiles).
// MODE 0: Q  -> [b][h][l][d] bf16, scaled by D^-1/2 * log2(e)
// MODE 1: K  -> [b][h][l][d] bf16
// MODE 2: V  -> [b][h][d][l] bf16 (transposed for PV B-fragments)
// MODE 3: G  -> sigmoid(x+bg)    f32 [m][n]
// MODE 4: out-> x+bo             f32 [m][n] (non-temporal store)
template <int MODE>
__global__ __launch_bounds__(256) void k_gemm(const bf16* __restrict__ A,
                                              const bf16* __restrict__ Wp,
                                              const float* __restrict__ bias1d,
                                              void* __restrict__ Out) {
  const int lane = threadIdx.x & 31;
  const int wave = threadIdx.x >> 5;
  const int hh   = lane >> 4;        // lane half
  const int r15  = lane & 15;
  const int m0   = (blockIdx.x >> 3) * 128 + (wave >> 1) * 32;
  const int n0   = (blockIdx.x & 7) * 128 + (wave & 1) * 64;
  const int nt0  = n0 >> 4;

  v8f acc[2][4] = {};

  for (int kt = 0; kt < 32; ++kt) {
    ABf a[2];
#pragma unroll
    for (int mi = 0; mi < 2; ++mi) {
      // A-fragment (16x32): elems 0-15 <-> k = hh*16+0..15
      const bf16* ap = A + (size_t)(m0 + mi * 16 + r15) * C_ + kt * 32 + hh * 16;
      a[mi].h[0] = *(const v8bf*)(ap);
      a[mi].h[1] = *(const v8bf*)(ap + 8);
    }
#pragma unroll
    for (int ni = 0; ni < 4; ++ni) {
      v16bf b = *(const v16bf*)(Wp + ((size_t)((nt0 + ni) * 32 + kt) * 32 + lane) * 16);
      acc[0][ni] = wmma_bf16(a[0].v, b, acc[0][ni]);
      acc[1][ni] = wmma_bf16(a[1].v, b, acc[1][ni]);
    }
  }

#pragma unroll
  for (int mi = 0; mi < 2; ++mi)
#pragma unroll
    for (int ni = 0; ni < 4; ++ni)
#pragma unroll
      for (int r = 0; r < 8; ++r) {
        float v = acc[mi][ni][r];
        int mr = m0 + mi * 16 + r + 8 * hh;   // C-layout row
        int nc = n0 + ni * 16 + r15;          // C-layout col
        if (MODE == 0 || MODE == 1) {
          if (MODE == 0) v *= 0.125f * LOG2E; // 1/sqrt(64), exp2 domain
          int b = mr >> 11, l = mr & (L_ - 1), h = nc >> 6, d = nc & (D_ - 1);
          ((bf16*)Out)[((size_t)(b * H_ + h) * L_ + l) * D_ + d] = (bf16)v;
        } else if (MODE == 2) {
          int b = mr >> 11, l = mr & (L_ - 1), h = nc >> 6, d = nc & (D_ - 1);
          ((bf16*)Out)[((size_t)(b * H_ + h) * D_ + d) * L_ + l] = (bf16)v;
        } else if (MODE == 3) {
          float g = 1.0f / (1.0f + __expf(-(v + bias1d[nc])));
          ((float*)Out)[(size_t)mr * C_ + nc] = g;
        } else {
          __builtin_nontemporal_store(v + bias1d[nc],
                                      &((float*)Out)[(size_t)mr * C_ + nc]);
        }
      }
}

// --------------------------- flash attention -------------------------------
// grid = B*H*(L/128); block = 8 waves; each wave owns 16 q rows, loops over
// kv in tiles of 32. Bias (the HBM-bound stream) is software-pipelined one
// tile ahead with non-temporal loads and consumed as the WMMA C accumulator.
// P relaid C->A layout via a per-wave LDS tile (DS in-order within a wave).
__global__ __launch_bounds__(256) void k_attn(const bf16* __restrict__ Qh,
                                              const bf16* __restrict__ Kh,
                                              const bf16* __restrict__ Vt,
                                              const float* __restrict__ Bias,
                                              const float* __restrict__ G,
                                              bf16* __restrict__ Yb) {
  const int lane = threadIdx.x & 31;
  const int wave = threadIdx.x >> 5;
  const int hh   = lane >> 4;
  const int r15  = lane & 15;
  const int qt   = blockIdx.x & 15;
  const int h    = (blockIdx.x >> 4) & 15;
  const int b    = blockIdx.x >> 8;
  const int q0   = qt * 128 + wave * 16;

  const bf16*  Qp = Qh + ((size_t)(b * H_ + h) * L_ + q0) * D_;
  const bf16*  Kp = Kh + (size_t)(b * H_ + h) * L_ * D_;
  const bf16*  Vp = Vt + (size_t)(b * H_ + h) * D_ * L_;
  const float* Bp = Bias + ((size_t)(b * H_ + h) * L_ + q0) * L_;

  // Q A-fragments, two 32-wide d chunks (scale*log2e folded into Qh)
  ABf qa[2];
#pragma unroll
  for (int c = 0; c < 2; ++c) {
    const bf16* qp = Qp + (size_t)r15 * D_ + c * 32 + hh * 16;
    qa[c].h[0] = *(const v8bf*)(qp);
    qa[c].h[1] = *(const v8bf*)(qp + 8);
  }

  v8f y[4] = {};
  float mrow[8], lrow[8];
#pragma unroll
  for (int r = 0; r < 8; ++r) { mrow[r] = -INFINITY; lrow[r] = 0.0f; }

  __shared__ __align__(32) bf16 pbuf[8][16][32];

  // prefetch bias tile 0 (raw, unscaled -> no wait at prefetch site)
  v8f bn[2];
#pragma unroll
  for (int t = 0; t < 2; ++t)
#pragma unroll
    for (int r = 0; r < 8; ++r)
      bn[t][r] = __builtin_nontemporal_load(
          Bp + (size_t)(r + 8 * hh) * L_ + t * 16 + r15);

  for (int kv = 0; kv < L_; kv += 32) {
    // consume prefetched bias into C accumulators (exp2 domain)
    v8f s[2];
#pragma unroll
    for (int t = 0; t < 2; ++t)
#pragma unroll
      for (int r = 0; r < 8; ++r) s[t][r] = bn[t][r] * LOG2E;

    // prefetch next tile's bias (wraps to 0 on last iter: harmless)
    const int kvn = (kv + 32) & (L_ - 1);
#pragma unroll
    for (int t = 0; t < 2; ++t)
#pragma unroll
      for (int r = 0; r < 8; ++r)
        bn[t][r] = __builtin_nontemporal_load(
            Bp + (size_t)(r + 8 * hh) * L_ + kvn + t * 16 + r15);

    // scores += q @ k^T   (K-dim = d, split 2x32)
#pragma unroll
    for (int c = 0; c < 2; ++c)
#pragma unroll
      for (int t = 0; t < 2; ++t) {
        v16bf kb = *(const v16bf*)(Kp + (size_t)(kv + t * 16 + r15) * D_ +
                                   c * 32 + hh * 16);
        s[t] = wmma_bf16(qa[c].v, kb, s[t]);
      }

    // online softmax (rows in VGPR index r, cols across 16-lane halves)
    float alpha[8];
#pragma unroll
    for (int r = 0; r < 8; ++r) {
      float v = xmax16(fmaxf(s[0][r], s[1][r]));
      float mnew = fmaxf(mrow[r], v);
      alpha[r] = __builtin_amdgcn_exp2f(mrow[r] - mnew);
      s[0][r]  = __builtin_amdgcn_exp2f(s[0][r] - mnew);
      s[1][r]  = __builtin_amdgcn_exp2f(s[1][r] - mnew);
      lrow[r]  = lrow[r] * alpha[r] + xsum16(s[0][r] + s[1][r]);
      mrow[r]  = mnew;
    }
#pragma unroll
    for (int d4 = 0; d4 < 4; ++d4)
#pragma unroll
      for (int r = 0; r < 8; ++r) y[d4][r] *= alpha[r];

    // P: C-layout -> A-layout via per-wave LDS tile
#pragma unroll
    for (int t = 0; t < 2; ++t)
#pragma unroll
      for (int r = 0; r < 8; ++r)
        pbuf[wave][r + 8 * hh][t * 16 + r15] = (bf16)s[t][r];
    asm volatile("" ::: "memory");
    ABf pa;
    pa.h[0] = *(const v8bf*)(&pbuf[wave][r15][hh * 16]);
    pa.h[1] = *(const v8bf*)(&pbuf[wave][r15][hh * 16 + 8]);
    asm volatile("" ::: "memory");

    // y += p @ v   (V stored [d][l]: B-fragment loads contiguous in l)
#pragma unroll
    for (int d4 = 0; d4 < 4; ++d4) {
      v16bf vb = *(const v16bf*)(Vp + (size_t)(d4 * 16 + r15) * L_ + kv + hh * 16);
      y[d4] = wmma_bf16(pa.v, vb, y[d4]);
    }
  }

  // normalize, gate, store bf16 Y for the final Wo GEMM
#pragma unroll
  for (int r = 0; r < 8; ++r) lrow[r] = 1.0f / lrow[r];
#pragma unroll
  for (int d4 = 0; d4 < 4; ++d4)
#pragma unroll
    for (int r = 0; r < 8; ++r) {
      int l  = q0 + r + 8 * hh;
      int cc = h * D_ + d4 * 16 + r15;
      size_t idx = ((size_t)b * L_ + l) * C_ + cc;
      Yb[idx] = (bf16)(y[d4][r] * lrow[r] * G[idx]);
    }
}

// ------------------------------- launcher ----------------------------------
extern "C" void kernel_launch(void* const* d_in, const int* in_sizes, int n_in,
                              void* d_out, int out_size, void* d_ws,
                              size_t ws_size, hipStream_t stream) {
  const float* query = (const float*)d_in[0];
  const float* bias  = (const float*)d_in[1];
  const float* Wq    = (const float*)d_in[2];
  const float* Wk    = (const float*)d_in[3];
  const float* Wv    = (const float*)d_in[4];
  const float* Wg    = (const float*)d_in[5];
  const float* bg    = (const float*)d_in[6];
  const float* Wo    = (const float*)d_in[7];
  const float* bo    = (const float*)d_in[8];
  float* out = (float*)d_out;

  char* ws = (char*)d_ws;
  const size_t MB = 1024 * 1024;
  bf16*  Xb  = (bf16*)(ws);            //  8 MB query bf16 [4096][1024]
  bf16*  Wpq = (bf16*)(ws +  8 * MB);  //  2 MB packed weights...
  bf16*  Wpk = (bf16*)(ws + 10 * MB);
  bf16*  Wpv = (bf16*)(ws + 12 * MB);
  bf16*  Wpg = (bf16*)(ws + 14 * MB);
  bf16*  Wpo = (bf16*)(ws + 16 * MB);
  bf16*  Qh  = (bf16*)(ws + 18 * MB);  //  8 MB [b][h][l][d]
  bf16*  Kh  = (bf16*)(ws + 26 * MB);  //  8 MB [b][h][l][d]
  bf16*  Vt  = (bf16*)(ws + 34 * MB);  //  8 MB [b][h][d][l]
  float* G   = (float*)(ws + 42 * MB); // 16 MB sigmoid gate f32
  bf16*  Yb  = (bf16*)(ws + 58 * MB);  //  8 MB gated attention output bf16

  k_cvt<<<(M_ * C_) / 256, 256, 0, stream>>>(query, Xb, M_ * C_);
  k_packw<<<(C_ * C_) / 256, 256, 0, stream>>>(Wq, Wpq);
  k_packw<<<(C_ * C_) / 256, 256, 0, stream>>>(Wk, Wpk);
  k_packw<<<(C_ * C_) / 256, 256, 0, stream>>>(Wv, Wpv);
  k_packw<<<(C_ * C_) / 256, 256, 0, stream>>>(Wg, Wpg);
  k_packw<<<(C_ * C_) / 256, 256, 0, stream>>>(Wo, Wpo);

  k_gemm<0><<<256, 256, 0, stream>>>(Xb, Wpq, nullptr, Qh);
  k_gemm<1><<<256, 256, 0, stream>>>(Xb, Wpk, nullptr, Kh);
  k_gemm<2><<<256, 256, 0, stream>>>(Xb, Wpv, nullptr, Vt);
  k_gemm<3><<<256, 256, 0, stream>>>(Xb, Wpg, bg, G);

  k_attn<<<B_ * H_ * (L_ / 128), 256, 0, stream>>>(Qh, Kh, Vt, bias, G, Yb);

  k_gemm<4><<<256, 256, 0, stream>>>(Yb, Wpo, bo, out);
}
